// MoEVisionBackbone_55705725829210
// MI455X (gfx1250) — matrix operand order repaired and checked by reading, not compiled, not run on hardware
//
#include <hip/hip_runtime.h>
#include <math.h>

// ---------------------------------------------------------------------------
// Types for CDNA5 WMMA (wave32, 16x16x32 bf16 -> f32)
// ---------------------------------------------------------------------------
typedef __attribute__((ext_vector_type(16))) __bf16    bf16x16;
typedef __attribute__((ext_vector_type(8)))  float     f32x8;
typedef __attribute__((ext_vector_type(4)))  unsigned  u32x4;
typedef __attribute__((ext_vector_type(2)))  unsigned  u32x2;

union Frag { bf16x16 v; u32x4 q[2]; };

__device__ __forceinline__ __bf16 f2bf(float f) {
  union { float f; unsigned u; } c; c.f = f;
  unsigned r = c.u + 0x7FFFu + ((c.u >> 16) & 1u);   // round-to-nearest-even
  union { unsigned short s; __bf16 b; } o; o.s = (unsigned short)(r >> 16);
  return o.b;
}

__device__ __forceinline__ unsigned pk2bf(float a, float b) {
  union { float f; unsigned u; } ca, cb; ca.f = a; cb.f = b;
  const unsigned ra = ca.u + 0x7FFFu + ((ca.u >> 16) & 1u);
  const unsigned rb = cb.u + 0x7FFFu + ((cb.u >> 16) & 1u);
  return (ra >> 16) | (rb & 0xFFFF0000u);
}

// Store 4 floats as 4 packed bf16 (8 bytes) to LDS.
__device__ __forceinline__ void store4bf(__bf16* p, float4 v) {
  u32x2 d;
  d.x = pk2bf(v.x, v.y);
  d.y = pk2bf(v.z, v.w);
  *reinterpret_cast<u32x2*>(p) = d;
}

__device__ __forceinline__ float gelu_erf(float x) {
  return 0.5f * x * (1.0f + erff(x * 0.7071067811865475f));
}

// Block-wide sum reduction, blockDim.x == 256, s has 256 floats.
__device__ __forceinline__ float block_reduce_sum(float v, float* s) {
  const int t = threadIdx.x;
  s[t] = v; __syncthreads();
  #pragma unroll
  for (int off = 128; off > 0; off >>= 1) {
    if (t < off) s[t] += s[t + off];
    __syncthreads();
  }
  float r = s[0]; __syncthreads();
  return r;
}

// ---------------------------------------------------------------------------
// WMMA fragment loaders (ISA 7.12.2 layouts), LDS row pitch = 40 halves (80B)
// ---------------------------------------------------------------------------
static constexpr int BK    = 32;
static constexpr int PITCH = 40;   // halves; 80B rows keep 16B alignment + bank skew

__device__ __forceinline__ void load_a_frag(Frag& f, const __bf16* lds_row_base, int lane) {
  // lane<16: row=lane,    K pairs 0..7 then 16..23  (byte off 0, 32)
  // lane>=16: row=lane-16, K pairs 8..15 then 24..31 (byte off 16, 48)
  const int ab = (lane < 16) ? 0 : 16;
  const char* p = reinterpret_cast<const char*>(lds_row_base);
  f.q[0] = *reinterpret_cast<const u32x4*>(p + ab);
  f.q[1] = *reinterpret_cast<const u32x4*>(p + 32 + ab);
}

__device__ __forceinline__ void load_b_frag(Frag& f, const __bf16* lds_row_base, int lane) {
  // lane<16: N=lane, K 0..7 then 16..23 ; lane>=16: N=lane-16, K 8..15 then 24..31 (byte off 16, 48)
  const int bb = (lane < 16) ? 0 : 16;
  const char* p = reinterpret_cast<const char*>(lds_row_base);
  f.q[0] = *reinterpret_cast<const u32x4*>(p + bb);
  f.q[1] = *reinterpret_cast<const u32x4*>(p + 32 + bb);
}

// ---------------------------------------------------------------------------
// Implicit-GEMM conv (VALID, stride S) with WMMA; all geometry compile-time.
//   M = NB*OH*OW (rows), N = CO (cols), K = CI*KH*KW. All divisible by tiles.
//   Output scattered directly to NCHW for the following BatchNorm.
// ---------------------------------------------------------------------------
template<int BM, int BN, int WAVES_M, int WAVES_N,
         int CI, int IH, int IW, int CO, int OH, int OW, int KH, int KW, int S>
__global__ __launch_bounds__(256)
void conv_wmma(const float* __restrict__ in, const float* __restrict__ w,
               const float* __restrict__ bias, float* __restrict__ out)
{
  constexpr int K   = CI * KH * KW;
  constexpr int KHW = KH * KW;
  constexpr int HWo = OH * OW;
  constexpr int TM  = BM / WAVES_M;
  constexpr int TN  = BN / WAVES_N;
  constexpr int FM  = TM / 16;
  constexpr int FN  = TN / 16;
  constexpr int AIT = (BM * BK) / 256;        // scalar-gather iterations for A
  constexpr int BIT = (BN * BK) / (4 * 256);  // float4 iterations for B

  __shared__ __align__(16) __bf16 lA[BM * PITCH];
  __shared__ __align__(16) __bf16 lB[BN * PITCH];

  const int tid  = threadIdx.x;
  const int lane = tid & 31;
  const int wv   = tid >> 5;
  const int wm   = wv % WAVES_M;
  const int wn   = wv / WAVES_M;
  const int m0   = blockIdx.y * BM;
  const int n0   = blockIdx.x * BN;

  // Hoisted per-thread im2col row bases (invariant over k0).
  const int ac = tid & 31;           // K offset within tile, fixed per thread
  int aBase[AIT];
  #pragma unroll
  for (int it = 0; it < AIT; ++it) {
    const int r   = (tid >> 5) + 8 * it;
    const int m   = m0 + r;
    const int n   = m / HWo;                 // compile-time divisors
    const int rem = m - n * HWo;
    const int oh  = rem / OW;
    const int ow  = rem - oh * OW;
    aBase[it] = ((n * CI) * IH + oh * S) * IW + ow * S;
  }

  f32x8 acc[FM][FN];
  #pragma unroll
  for (int i = 0; i < FM; ++i)
    #pragma unroll
    for (int j = 0; j < FN; ++j)
      acc[i][j] = f32x8{0.f,0.f,0.f,0.f,0.f,0.f,0.f,0.f};

  for (int k0 = 0; k0 < K; k0 += BK) {
    // A tile: im2col gather, f32 -> bf16 (constant-divisor decode of k)
    {
      const int k  = k0 + ac;
      const int ci = k / KHW;
      const int kr = k - ci * KHW;
      const int kh = kr / KW;
      const int kw = kr - kh * KW;
      const int ko = ci * (IH * IW) + kh * IW + kw;
      #pragma unroll
      for (int it = 0; it < AIT; ++it) {
        const int r = (tid >> 5) + 8 * it;
        lA[r * PITCH + ac] = f2bf(in[aBase[it] + ko]);
      }
    }
    // B tile: weights [CO][K] row-major, vectorized float4 -> packed bf16x4
    #pragma unroll
    for (int it = 0; it < BIT; ++it) {
      const int i  = tid + 256 * it;
      const int r  = i >> 3;
      const int c4 = (i & 7) << 2;
      const float4 v = *reinterpret_cast<const float4*>(&w[(r + n0) * K + k0 + c4]);
      store4bf(&lB[r * PITCH + c4], v);
    }
    // Prefetch next weight tile into cache while WMMAs run (uniform branch).
    if (k0 + BK < K) {
      const int r  = tid >> 3;
      const int c4 = (tid & 7) << 2;
      if (r < BN) __builtin_prefetch(&w[(r + n0) * K + k0 + BK + c4], 0, 3);
    }
    __syncthreads();

    Frag a[FM], b[FN];
    #pragma unroll
    for (int i = 0; i < FM; ++i)
      load_a_frag(a[i], &lA[(wm * TM + i * 16 + (lane & 15)) * PITCH], lane);
    #pragma unroll
    for (int j = 0; j < FN; ++j)
      load_b_frag(b[j], &lB[(wn * TN + j * 16 + (lane & 15)) * PITCH], lane);

    #pragma unroll
    for (int i = 0; i < FM; ++i)
      #pragma unroll
      for (int j = 0; j < FN; ++j)
        acc[i][j] = __builtin_amdgcn_wmma_f32_16x16x32_bf16(
            false, a[i].v, false, b[j].v, (short)0, acc[i][j], false, false);
    __syncthreads();
  }

  // Epilogue: +bias, scatter to NCHW (compile-time divisors)
  #pragma unroll
  for (int i = 0; i < FM; ++i) {
    #pragma unroll
    for (int j = 0; j < FN; ++j) {
      const int col  = n0 + wn * TN + j * 16 + (lane & 15);
      const float bv = bias[col];
      const int rb   = m0 + wm * TM + i * 16 + ((lane >> 4) << 3);
      #pragma unroll
      for (int e = 0; e < 8; ++e) {
        const int m   = rb + e;
        const int n   = m / HWo;
        const int rem = m - n * HWo;
        const int oh  = rem / OW;
        const int ow  = rem - oh * OW;
        out[((n * CO + col) * OH + oh) * OW + ow] = acc[i][j][e] + bv;
      }
    }
  }
}

// ---------------------------------------------------------------------------
// Dense WMMA GEMM: C = A[M,K] * W[N,K]^T + bias.
// EPI: 0 = store, 1 = store+gelu, 2 = C += cs*rowScale[m]*(val+bias)
// ---------------------------------------------------------------------------
template<int BM, int BN, int WAVES_M, int WAVES_N, int EPI>
__global__ __launch_bounds__(256)
void gemm_wmma(const float* __restrict__ A, const float* __restrict__ W,
               const float* __restrict__ bias, float* __restrict__ C,
               int K, int ldc, float constScale,
               const float* __restrict__ rowScale, int rsStride)
{
  constexpr int TM  = BM / WAVES_M;
  constexpr int TN  = BN / WAVES_N;
  constexpr int FM  = TM / 16;
  constexpr int FN  = TN / 16;
  constexpr int AIT = (BM * BK) / (4 * 256);  // float4 iterations
  constexpr int BIT = (BN * BK) / (4 * 256);

  __shared__ __align__(16) __bf16 lA[BM * PITCH];
  __shared__ __align__(16) __bf16 lB[BN * PITCH];

  const int tid  = threadIdx.x;
  const int lane = tid & 31;
  const int wv   = tid >> 5;
  const int wm   = wv % WAVES_M;
  const int wn   = wv / WAVES_M;
  const int m0   = blockIdx.y * BM;
  const int n0   = blockIdx.x * BN;

  f32x8 acc[FM][FN];
  #pragma unroll
  for (int i = 0; i < FM; ++i)
    #pragma unroll
    for (int j = 0; j < FN; ++j)
      acc[i][j] = f32x8{0.f,0.f,0.f,0.f,0.f,0.f,0.f,0.f};

  for (int k0 = 0; k0 < K; k0 += BK) {
    #pragma unroll
    for (int it = 0; it < AIT; ++it) {
      const int i  = tid + 256 * it;
      const int r  = i >> 3;
      const int c4 = (i & 7) << 2;
      const float4 v = *reinterpret_cast<const float4*>(&A[(size_t)(m0 + r) * K + k0 + c4]);
      store4bf(&lA[r * PITCH + c4], v);
    }
    #pragma unroll
    for (int it = 0; it < BIT; ++it) {
      const int i  = tid + 256 * it;
      const int r  = i >> 3;
      const int c4 = (i & 7) << 2;
      const float4 v = *reinterpret_cast<const float4*>(&W[(size_t)(n0 + r) * K + k0 + c4]);
      store4bf(&lB[r * PITCH + c4], v);
    }
    // Prefetch next K tile (uniform branch; EXEC untouched).
    if (k0 + BK < K) {
      const int r  = tid >> 3;
      const int c4 = (tid & 7) << 2;
      __builtin_prefetch(&A[(size_t)(m0 + r) * K + k0 + BK + c4], 0, 3);
      __builtin_prefetch(&W[(size_t)(n0 + (r & (BN - 1))) * K + k0 + BK + c4], 0, 3);
    }
    __syncthreads();

    Frag a[FM], b[FN];
    #pragma unroll
    for (int i = 0; i < FM; ++i)
      load_a_frag(a[i], &lA[(wm * TM + i * 16 + (lane & 15)) * PITCH], lane);
    #pragma unroll
    for (int j = 0; j < FN; ++j)
      load_b_frag(b[j], &lB[(wn * TN + j * 16 + (lane & 15)) * PITCH], lane);

    #pragma unroll
    for (int i = 0; i < FM; ++i)
      #pragma unroll
      for (int j = 0; j < FN; ++j)
        acc[i][j] = __builtin_amdgcn_wmma_f32_16x16x32_bf16(
            false, a[i].v, false, b[j].v, (short)0, acc[i][j], false, false);
    __syncthreads();
  }

  #pragma unroll
  for (int i = 0; i < FM; ++i) {
    #pragma unroll
    for (int j = 0; j < FN; ++j) {
      const int col  = n0 + wn * TN + j * 16 + (lane & 15);
      const float bv = bias[col];
      const int rb   = m0 + wm * TM + i * 16 + ((lane >> 4) << 3);
      #pragma unroll
      for (int e = 0; e < 8; ++e) {
        const int m = rb + e;
        float v = acc[i][j][e] + bv;
        float* dst = &C[(size_t)m * ldc + col];
        if (EPI == 0) {
          *dst = v;
        } else if (EPI == 1) {
          *dst = gelu_erf(v);
        } else {
          float s = constScale;
          if (rowScale) s *= rowScale[(size_t)m * rsStride];
          *dst = *dst + s * v;
        }
      }
    }
  }
}

// ---------------------------------------------------------------------------
// BatchNorm (training mode) stats + apply(+gelu), LayerNorm, router, losses
// ---------------------------------------------------------------------------
__global__ __launch_bounds__(256)
void bn_stats(const float* __restrict__ x, float* __restrict__ mean,
              float* __restrict__ inv, int C, int NB, int HW)
{
  __shared__ float s[256];
  const int c = blockIdx.x;
  float s1 = 0.f, s2 = 0.f;
  for (int n = 0; n < NB; ++n) {
    const float* base = x + ((size_t)n * C + c) * HW;
    for (int p = threadIdx.x; p < HW; p += 256) {
      const float v = base[p];
      s1 += v; s2 += v * v;
    }
  }
  const float S1 = block_reduce_sum(s1, s);
  const float S2 = block_reduce_sum(s2, s);
  if (threadIdx.x == 0) {
    const float plane = (float)NB * (float)HW;
    const float m  = S1 / plane;
    const float vr = S2 / plane - m * m;
    mean[c] = m;
    inv[c]  = rsqrtf(vr + 1e-5f);
  }
}

// grid.x = NB*C planes, grid.y covers HW; channel decode is per-block scalar.
__global__ __launch_bounds__(256)
void bn_apply_gelu(float* __restrict__ x, const float* __restrict__ mean,
                   const float* __restrict__ inv, const float* __restrict__ g,
                   const float* __restrict__ b, int C, int HW)
{
  const int plane = blockIdx.x;
  const int c     = plane % C;
  const int p     = blockIdx.y * 256 + threadIdx.x;
  if (p >= HW) return;
  float* base = x + (size_t)plane * HW;
  const float v = base[p];
  base[p] = gelu_erf(g[c] * (v - mean[c]) * inv[c] + b[c]);
}

template<bool GELU_OUT>
__global__ __launch_bounds__(256)
void ln_rows(const float* __restrict__ x, const float* __restrict__ g,
             const float* __restrict__ bb, float* __restrict__ y, int D)
{
  __shared__ float s[256];
  const int b = blockIdx.x, t = threadIdx.x;
  const float v    = x[(size_t)b * D + t];
  const float mean = block_reduce_sum(v, s) / (float)D;
  const float d    = v - mean;
  const float var  = block_reduce_sum(d * d, s) / (float)D;
  float o = g[t] * d * rsqrtf(var + 1e-5f) + bb[t];
  if (GELU_OUT) o = gelu_erf(o);
  y[(size_t)b * D + t] = o;
}

__global__ __launch_bounds__(256)
void router_kernel(const float* __restrict__ h, const float* __restrict__ rw,
                   const float* __restrict__ ebias, float* __restrict__ combine,
                   float* __restrict__ probsum, float* __restrict__ zsum,
                   int B, int D)
{
  const int b = blockIdx.x * 256 + threadIdx.x;
  if (b >= B) return;
  float logits[8];
  #pragma unroll
  for (int e = 0; e < 8; ++e) logits[e] = ebias[e];
  for (int d = 0; d < D; ++d) {
    const float hv = h[(size_t)b * D + d];
    #pragma unroll
    for (int e = 0; e < 8; ++e) logits[e] += hv * rw[e * D + d];
  }
  float mx = logits[0];
  #pragma unroll
  for (int e = 1; e < 8; ++e) mx = fmaxf(mx, logits[e]);
  float p[8], sum = 0.f;
  #pragma unroll
  for (int e = 0; e < 8; ++e) { p[e] = expf(logits[e] - mx); sum += p[e]; }
  #pragma unroll
  for (int e = 0; e < 8; ++e) p[e] /= sum;
  // top-2
  int i1 = 0;
  #pragma unroll
  for (int e = 1; e < 8; ++e) if (p[e] > p[i1]) i1 = e;
  int i2 = (i1 == 0) ? 1 : 0;
  #pragma unroll
  for (int e = 0; e < 8; ++e) if (e != i1 && p[e] > p[i2]) i2 = e;
  const float tot = p[i1] + p[i2] + 1e-8f;
  #pragma unroll
  for (int e = 0; e < 8; ++e) combine[(size_t)b * 8 + e] = 0.f;
  combine[(size_t)b * 8 + i1] = p[i1] / tot;
  combine[(size_t)b * 8 + i2] = p[i2] / tot;
  float zs = 0.f;
  #pragma unroll
  for (int e = 0; e < 8; ++e) zs += logits[e] * logits[e];
  atomicAdd(zsum, zs);
  #pragma unroll
  for (int e = 0; e < 8; ++e) atomicAdd(&probsum[e], p[e]);
}

__global__ void losses_kernel(const float* __restrict__ probsum,
                              const float* __restrict__ zsum,
                              float* __restrict__ out_tail, int B)
{
  if (threadIdx.x == 0 && blockIdx.x == 0) {
    float lb = 0.f;
    #pragma unroll
    for (int e = 0; e < 8; ++e) {
      const float pm = probsum[e] / (float)B - 0.125f;
      lb += pm * pm;
    }
    out_tail[0] = lb * 8.0f;
    out_tail[1] = zsum[0] / (float)(B * 8) * 0.001f;
  }
}

__global__ __launch_bounds__(256)
void zero_f32(float* __restrict__ p, int n)
{
  const int i = blockIdx.x * 256 + threadIdx.x;
  if (i < n) p[i] = 0.f;
}

// ---------------------------------------------------------------------------
// Host-side launch
// ---------------------------------------------------------------------------
extern "C" void kernel_launch(void* const* d_in, const int* in_sizes, int n_in,
                              void* d_out, int out_size, void* d_ws, size_t ws_size,
                              hipStream_t stream)
{
  (void)in_sizes; (void)n_in; (void)out_size; (void)ws_size;

  const float* x       = (const float*)d_in[0];
  const float* c1w     = (const float*)d_in[1];
  const float* c1b     = (const float*)d_in[2];
  const float* bn1g    = (const float*)d_in[3];
  const float* bn1b    = (const float*)d_in[4];
  const float* c2w     = (const float*)d_in[5];
  const float* c2b     = (const float*)d_in[6];
  const float* bn2g    = (const float*)d_in[7];
  const float* bn2b    = (const float*)d_in[8];
  const float* c3w     = (const float*)d_in[9];
  const float* c3b     = (const float*)d_in[10];
  const float* bn3g    = (const float*)d_in[11];
  const float* bn3b    = (const float*)d_in[12];
  const float* proj_w  = (const float*)d_in[13];
  const float* proj_b  = (const float*)d_in[14];
  const float* ln_g    = (const float*)d_in[15];
  const float* ln_b    = (const float*)d_in[16];
  const float* rw      = (const float*)d_in[17];
  const float* ebias   = (const float*)d_in[18];
  const float* ew1     = (const float*)d_in[19];
  const float* eb1     = (const float*)d_in[20];
  const float* ew2     = (const float*)d_in[21];
  const float* eb2     = (const float*)d_in[22];
  const float* sw1     = (const float*)d_in[23];
  const float* sb1     = (const float*)d_in[24];
  const float* sw2     = (const float*)d_in[25];
  const float* sb2     = (const float*)d_in[26];
  const float* mlng    = (const float*)d_in[27];
  const float* mlnb    = (const float*)d_in[28];

  constexpr int B = 1024, D = 256, E = 8, H = 512;
  constexpr int M1 = 1024 * 35 * 39;   // 1,397,760
  constexpr int M2 = 1024 * 16 * 18;   //   294,912
  constexpr int M3 = 1024 * 14 * 16;   //   229,376
  constexpr int T1 = M1 * 32;
  constexpr int T2 = M2 * 64;
  constexpr int T3 = M3 * 128;

  // Workspace layout (floats); conv3 output reuses conv1 buffer.
  float* w0 = (float*)d_ws;
  constexpr size_t O_BUF1 = 0;
  constexpr size_t O_BUF2 = O_BUF1 + (size_t)T1;
  constexpr size_t O_H    = O_BUF2 + (size_t)T2;
  constexpr size_t O_SH   = O_H    + (size_t)(B * D);
  constexpr size_t O_HID  = O_SH   + (size_t)(B * D);
  constexpr size_t O_MOE  = O_HID  + (size_t)(B * H);
  constexpr size_t O_CMB  = O_MOE  + (size_t)(B * D);
  constexpr size_t O_MEAN = O_CMB  + (size_t)(B * E);
  constexpr size_t O_INV  = O_MEAN + 128;
  constexpr size_t O_PS   = O_INV  + 128;

  float* buf1   = w0 + O_BUF1;
  float* buf2   = w0 + O_BUF2;
  float* hbuf   = w0 + O_H;
  float* shid   = w0 + O_SH;
  float* hid    = w0 + O_HID;
  float* moe    = w0 + O_MOE;
  float* cmb    = w0 + O_CMB;
  float* bmean  = w0 + O_MEAN;
  float* binv   = w0 + O_INV;
  float* psum   = w0 + O_PS;       // [8] probsum, then zsum
  float* zsum   = psum + 8;
  float* out    = (float*)d_out;

  // ---- conv1: [1024,4,144,160] -> [1024,32,35,39], K=256 -------------------
  conv_wmma<128, 32, 8, 1,  4, 144, 160, 32, 35, 39, 8, 8, 4>
      <<<dim3(1, M1 / 128), 256, 0, stream>>>(x, c1w, c1b, buf1);
  bn_stats<<<32, 256, 0, stream>>>(buf1, bmean, binv, 32, 1024, 35 * 39);
  bn_apply_gelu<<<dim3(1024 * 32, (35 * 39 + 255) / 256), 256, 0, stream>>>(
      buf1, bmean, binv, bn1g, bn1b, 32, 35 * 39);

  // ---- conv2: -> [1024,64,16,18], K=512 ------------------------------------
  conv_wmma<128, 64, 4, 2,  32, 35, 39, 64, 16, 18, 4, 4, 2>
      <<<dim3(1, M2 / 128), 256, 0, stream>>>(buf1, c2w, c2b, buf2);
  bn_stats<<<64, 256, 0, stream>>>(buf2, bmean, binv, 64, 1024, 16 * 18);
  bn_apply_gelu<<<dim3(1024 * 64, (16 * 18 + 255) / 256), 256, 0, stream>>>(
      buf2, bmean, binv, bn2g, bn2b, 64, 16 * 18);

  // ---- conv3: -> [1024,128,14,16] (reuses buf1), K=576 ---------------------
  conv_wmma<128, 64, 4, 2,  64, 16, 18, 128, 14, 16, 3, 3, 1>
      <<<dim3(2, M3 / 128), 256, 0, stream>>>(buf2, c3w, c3b, buf1);
  bn_stats<<<128, 256, 0, stream>>>(buf1, bmean, binv, 128, 1024, 14 * 16);
  bn_apply_gelu<<<dim3(1024 * 128, (14 * 16 + 255) / 256), 256, 0, stream>>>(
      buf1, bmean, binv, bn3g, bn3b, 128, 14 * 16);

  // ---- projection: [1024,28672] @ [256,28672]^T -> LN -> gelu --------------
  gemm_wmma<128, 64, 4, 2, 0><<<dim3(D / 64, B / 128), 256, 0, stream>>>(
      buf1, proj_w, proj_b, hbuf, 28672, D, 1.0f, nullptr, 0);
  ln_rows<true><<<B, 256, 0, stream>>>(hbuf, ln_g, ln_b, hbuf, D);

  // ---- router + zero accumulators ------------------------------------------
  zero_f32<<<(B * D + 255) / 256, 256, 0, stream>>>(moe, B * D);
  zero_f32<<<1, 256, 0, stream>>>(psum, 9);
  router_kernel<<<B / 256, 256, 0, stream>>>(hbuf, rw, ebias, cmb, psum, zsum, B, D);

  // ---- shared expert: moe += 0.5 * (gelu(h@sw1^T+sb1)@sw2^T + sb2) ---------
  gemm_wmma<128, 64, 4, 2, 1><<<dim3(D / 64, B / 128), 256, 0, stream>>>(
      hbuf, sw1, sb1, shid, D, D, 1.0f, nullptr, 0);
  gemm_wmma<128, 64, 4, 2, 2><<<dim3(D / 64, B / 128), 256, 0, stream>>>(
      shid, sw2, sb2, moe, D, D, 0.5f, nullptr, 0);

  // ---- experts (dense over all 8, combined with top-2 weights) -------------
  for (int e = 0; e < E; ++e) {
    gemm_wmma<128, 64, 4, 2, 1><<<dim3(H / 64, B / 128), 256, 0, stream>>>(
        hbuf, ew1 + (size_t)e * H * D, eb1 + (size_t)e * H, hid, D, H, 1.0f, nullptr, 0);
    gemm_wmma<128, 64, 4, 2, 2><<<dim3(D / 64, B / 128), 256, 0, stream>>>(
        hid, ew2 + (size_t)e * D * H, eb2 + (size_t)e * D, moe, H, D, 1.0f, cmb + e, E);
  }

  // ---- final LayerNorm -> d_out, then aux losses ---------------------------
  ln_rows<false><<<B, 256, 0, stream>>>(moe, mlng, mlnb, out, D);
  losses_kernel<<<1, 32, 0, stream>>>(psum, zsum, out + (size_t)B * D, B);
}
